// Faster_RCNN_67662914782056
// MI455X (gfx1250) — matrix operand order repaired and checked by reading, not compiled, not run on hardware
//
#include <hip/hip_runtime.h>

typedef __attribute__((ext_vector_type(16))) __bf16 v16bf;
typedef __attribute__((ext_vector_type(8)))  float  v8f;

union FragU { uint4 u[2]; v16bf v; };

static __device__ __forceinline__ unsigned short f2bf(float f) {
  unsigned int u = __float_as_uint(f);
  u += 0x7FFFu + ((u >> 16) & 1u);
  return (unsigned short)(u >> 16);
}

// Async copy global -> LDS (CDNA5), tracked by ASYNCcnt.
static __device__ __forceinline__ void async_b128(unsigned lds_byte_addr, const void* gaddr) {
  asm volatile("global_load_async_to_lds_b128 %0, %1, off"
               :: "v"(lds_byte_addr), "v"(gaddr)
               : "memory");
}

// ---------------------------------------------------------------------------
// 1) Weight transpose + convert: W[3,3,1024,2048] f32 -> Wt[2048][9216] bf16
// ---------------------------------------------------------------------------
__global__ void __launch_bounds__(256) wconv_kernel(const float* __restrict__ W,
                                                    unsigned short* __restrict__ Wt) {
  __shared__ float tile[32][33];
  const int ocb = blockIdx.x * 32;
  const int cb  = blockIdx.y * 32;
  const int t   = blockIdx.z;            // tap 0..8
  const int col = threadIdx.x & 31;
  const int r0  = threadIdx.x >> 5;      // 0..7
#pragma unroll
  for (int i = 0; i < 4; ++i) {
    const int row = r0 + i * 8;          // c within tile
    tile[row][col] = W[((size_t)(t * 1024 + cb + row) * 2048) + ocb + col];
  }
  __syncthreads();
#pragma unroll
  for (int i = 0; i < 4; ++i) {
    const int ocr = r0 + i * 8;          // oc within tile
    Wt[(size_t)(ocb + ocr) * 9216 + t * 1024 + cb + col] = f2bf(tile[col][ocr]);
  }
}

// ---------------------------------------------------------------------------
// 2) crop_and_resize (bilinear, TF semantics) + 2x2 maxpool, f32 -> bf16
//    out: roif[256][7][7][1024]
// ---------------------------------------------------------------------------
__global__ void __launch_bounds__(256) crop_pool_kernel(const float* __restrict__ fmap,
                                                        const float* __restrict__ boxes,
                                                        unsigned short* __restrict__ roif) {
  const int pix = blockIdx.x;            // 0..48
  const int n   = blockIdx.y;            // roi
  const int pi = pix / 7, pj = pix % 7;
  const int c = threadIdx.x * 4;         // 4 channels per thread
  const float y1 = boxes[n * 4 + 0], x1 = boxes[n * 4 + 1];
  const float y2 = boxes[n * 4 + 2], x2 = boxes[n * 4 + 3];
  float bx = -1e30f, by = -1e30f, bz = -1e30f, bw = -1e30f;
#pragma unroll
  for (int sy = 0; sy < 2; ++sy) {
    const int cy = 2 * pi + sy;
    const float ys = (y1 + ((float)cy * (1.0f / 13.0f)) * (y2 - y1)) * 63.0f;
    float y0f = fminf(fmaxf(floorf(ys), 0.0f), 63.0f);
    const float wy = ys - y0f;
    const int y0i = (int)y0f;
    const int y1i = (int)fminf(fmaxf(y0f + 1.0f, 0.0f), 63.0f);
#pragma unroll
    for (int sx = 0; sx < 2; ++sx) {
      const int cx = 2 * pj + sx;
      const float xs = (x1 + ((float)cx * (1.0f / 13.0f)) * (x2 - x1)) * 63.0f;
      float x0f = fminf(fmaxf(floorf(xs), 0.0f), 63.0f);
      const float wx = xs - x0f;
      const int x0i = (int)x0f;
      const int x1i = (int)fminf(fmaxf(x0f + 1.0f, 0.0f), 63.0f);
      const float4 v00 = *(const float4*)&fmap[((size_t)(y0i * 64 + x0i)) * 1024 + c];
      const float4 v01 = *(const float4*)&fmap[((size_t)(y0i * 64 + x1i)) * 1024 + c];
      const float4 v10 = *(const float4*)&fmap[((size_t)(y1i * 64 + x0i)) * 1024 + c];
      const float4 v11 = *(const float4*)&fmap[((size_t)(y1i * 64 + x1i)) * 1024 + c];
      const float wx0 = 1.0f - wx, wy0 = 1.0f - wy;
      const float tx0 = wx0 * v00.x + wx * v01.x, bx0 = wx0 * v10.x + wx * v11.x;
      const float ty0 = wx0 * v00.y + wx * v01.y, by0 = wx0 * v10.y + wx * v11.y;
      const float tz0 = wx0 * v00.z + wx * v01.z, bz0 = wx0 * v10.z + wx * v11.z;
      const float tw0 = wx0 * v00.w + wx * v01.w, bw0 = wx0 * v10.w + wx * v11.w;
      bx = fmaxf(bx, wy0 * tx0 + wy * bx0);
      by = fmaxf(by, wy0 * ty0 + wy * by0);
      bz = fmaxf(bz, wy0 * tz0 + wy * bz0);
      bw = fmaxf(bw, wy0 * tw0 + wy * bw0);
    }
  }
  unsigned short* dst = &roif[((size_t)(n * 49 + pix)) * 1024 + c];
  dst[0] = f2bf(bx); dst[1] = f2bf(by); dst[2] = f2bf(bz); dst[3] = f2bf(bw);
}

// ---------------------------------------------------------------------------
// 3) Conv 3x3 SAME as implicit GEMM (M=64 padded from 49, N=128 oc tile,
//    K=9216 = 9 taps x 1024 ch) with bf16 WMMA. Async double-buffered LDS
//    staging (GLOBAL_LOAD_ASYNC_TO_LDS_B128 + s_wait_asynccnt). Fused
//    bias + ReLU + spatial mean -> fc[256][2048].
// ---------------------------------------------------------------------------
__global__ void __launch_bounds__(256) conv_gemm_kernel(const unsigned short* __restrict__ roif,
                                                        const unsigned short* __restrict__ Wt,
                                                        const float* __restrict__ bias,
                                                        float* __restrict__ fc) {
  __shared__ __align__(16) unsigned short lds_a[2 * 64 * 32];   // [buf][m][k]
  __shared__ __align__(16) unsigned short lds_b[2 * 128 * 32];  // [buf][n][k]
  __shared__ __align__(16) unsigned short lds_dummy[8];         // sink for padded lanes
  __shared__ float lds_fc[64 * 128];                            // [pos][oc]

  const int ocb  = blockIdx.x * 128;
  const int n    = blockIdx.y;
  const int tid  = threadIdx.x;
  const int wave = tid >> 5;
  const int lane = tid & 31;
  const int g    = lane >> 4;          // half-wave group
  const int ln16 = lane & 15;
  const int mtile = wave & 3;          // 16-row M tile
  const int nq    = (wave >> 2) * 4;   // first of four 16-col N subtiles

  // A cooperative staging: one 16B chunk/thread (64 rows x 4 chunks)
  const int arow = tid >> 2;           // 0..63 = output position (padded)
  const int aq   = (tid & 3) * 8;      // half offset within 32-K chunk
  const int pi2 = arow / 7, pj2 = arow % 7;
  // B cooperative staging: two 16B chunks/thread (128 rows x 4 chunks)
  const int brow = tid >> 1;           // 0..127 = oc row
  const int bq   = (tid & 1) * 16;     // half offset (chunks at bq, bq+8)

  const size_t roi_base = (size_t)n * 49 * 1024;
  const size_t wt_base  = (size_t)(ocb + brow) * 9216 + bq;

  const unsigned a_slot  = (unsigned)(size_t)&lds_a[arow * 32 + aq];
  const unsigned b_slot0 = (unsigned)(size_t)&lds_b[brow * 32 + bq];
  const unsigned dum     = (unsigned)(size_t)&lds_dummy[0];

  v8f acc0 = {}, acc1 = {}, acc2 = {}, acc3 = {};

  // Issue the 3 async transfers (1 A chunk + 2 B chunks) for K-step kt into buf
  auto issue = [&](int kt, int buf) {
    const int tap = kt >> 5;                 // 32 K-chunks per tap
    const int c0  = (kt & 31) << 5;
    const int t3  = tap / 3;
    const int dy = t3 - 1, dx = (tap - t3 * 3) - 1;
    const int yy = pi2 + dy, xx = pj2 + dx;
    const bool aval = (arow < 49) && ((unsigned)yy < 7u) && ((unsigned)xx < 7u);
    const unsigned short* agv = &roif[roi_base + (size_t)(yy * 7 + xx) * 1024 + c0 + aq];
    const unsigned short* ag  = aval ? agv : roif;             // safe address
    const unsigned adst = aval ? (a_slot + (unsigned)buf * 4096u) : dum;
    async_b128(adst, ag);                                      // uniform: all lanes
    if (!aval) {                                               // SAME-padding zeros
      *(uint4*)&lds_a[buf * 2048 + arow * 32 + aq] = make_uint4(0u, 0u, 0u, 0u);
    }
    const unsigned short* bg = &Wt[wt_base + ((size_t)tap << 10) + c0];
    async_b128(b_slot0 + (unsigned)buf * 8192u, bg);
    async_b128(b_slot0 + (unsigned)buf * 8192u + 16u, bg + 8);
  };

  issue(0, 0);  // prologue

  for (int kt = 0; kt < 288; ++kt) {
    const int cur = kt & 1;
    if (kt + 1 < 288) {
      issue(kt + 1, cur ^ 1);                                  // overlap next tile
      asm volatile("s_wait_asynccnt 0x3" ::: "memory");        // prev tile complete
    } else {
      asm volatile("s_wait_asynccnt 0x0" ::: "memory");
    }
    __syncthreads();

    // Fragments: contiguous K order (K = 16g + s) in both A(16x32) and B(32x16)
    const unsigned short* abase = lds_a + cur * 2048;
    const unsigned short* bbase = lds_b + cur * 4096;
    FragU fa;
    const unsigned short* ap = abase + (mtile * 16 + ln16) * 32;
    fa.u[0] = *(const uint4*)&ap[16 * g];         // K = 16g..16g+7
    fa.u[1] = *(const uint4*)&ap[16 * g + 8];     // K = 16g+8..16g+15
    FragU fb0, fb1, fb2, fb3;
    const unsigned short* bp0 = bbase + ((nq + 0) * 16 + ln16) * 32 + 16 * g;
    const unsigned short* bp1 = bbase + ((nq + 1) * 16 + ln16) * 32 + 16 * g;
    const unsigned short* bp2 = bbase + ((nq + 2) * 16 + ln16) * 32 + 16 * g;
    const unsigned short* bp3 = bbase + ((nq + 3) * 16 + ln16) * 32 + 16 * g;
    fb0.u[0] = *(const uint4*)&bp0[0]; fb0.u[1] = *(const uint4*)&bp0[8];
    fb1.u[0] = *(const uint4*)&bp1[0]; fb1.u[1] = *(const uint4*)&bp1[8];
    fb2.u[0] = *(const uint4*)&bp2[0]; fb2.u[1] = *(const uint4*)&bp2[8];
    fb3.u[0] = *(const uint4*)&bp3[0]; fb3.u[1] = *(const uint4*)&bp3[8];

    acc0 = __builtin_amdgcn_wmma_f32_16x16x32_bf16(false, fa.v, false, fb0.v,
                                                   (short)0, acc0, false, false);
    acc1 = __builtin_amdgcn_wmma_f32_16x16x32_bf16(false, fa.v, false, fb1.v,
                                                   (short)0, acc1, false, false);
    acc2 = __builtin_amdgcn_wmma_f32_16x16x32_bf16(false, fa.v, false, fb2.v,
                                                   (short)0, acc2, false, false);
    acc3 = __builtin_amdgcn_wmma_f32_16x16x32_bf16(false, fa.v, false, fb3.v,
                                                   (short)0, acc3, false, false);
    __syncthreads();   // all reads of `cur` done before it is refilled
  }

  // Epilogue: scatter accumulators, then fused bias + ReLU + mean over 49 pos
  {
    const int nc0 = (nq + 0) * 16 + ln16;
    const int nc1 = (nq + 1) * 16 + ln16;
    const int nc2 = (nq + 2) * 16 + ln16;
    const int nc3 = (nq + 3) * 16 + ln16;
#pragma unroll
    for (int r = 0; r < 8; ++r) {
      const int pos = mtile * 16 + r + 8 * g;   // C/D layout: VGPR r -> M=r+8g
      lds_fc[pos * 128 + nc0] = acc0[r];
      lds_fc[pos * 128 + nc1] = acc1[r];
      lds_fc[pos * 128 + nc2] = acc2[r];
      lds_fc[pos * 128 + nc3] = acc3[r];
    }
  }
  __syncthreads();
  if (tid < 128) {
    const float b = bias[ocb + tid];
    float s = 0.0f;
    for (int p = 0; p < 49; ++p) {
      const float v = lds_fc[p * 128 + tid] + b;
      s += v > 0.0f ? v : 0.0f;
    }
    fc[(size_t)n * 2048 + ocb + tid] = s * (1.0f / 49.0f);
  }
}

// ---------------------------------------------------------------------------
// 4) FC heads: score = fc@Wc + bc, bbox = fc@Wb + bb (tiny; fp32, coalesced N)
// ---------------------------------------------------------------------------
__global__ void __launch_bounds__(512) fc_kernel(const float* __restrict__ fc,
                                                 const float* __restrict__ Wc,
                                                 const float* __restrict__ bc,
                                                 const float* __restrict__ Wb,
                                                 const float* __restrict__ bb,
                                                 float* __restrict__ score,
                                                 float* __restrict__ bbox) {
  __shared__ float row[2048];
  const int n = blockIdx.x;
  for (int i = threadIdx.x; i < 2048; i += 512) row[i] = fc[(size_t)n * 2048 + i];
  __syncthreads();
  const int t = threadIdx.x;
  if (t < 81) {
    float s = 0.0f;
    for (int k = 0; k < 2048; ++k) s = fmaf(row[k], Wc[(size_t)k * 81 + t], s);
    score[n * 81 + t] = s + bc[t];
  } else if (t < 405) {
    const int j = t - 81;
    float s = 0.0f;
    for (int k = 0; k < 2048; ++k) s = fmaf(row[k], Wb[(size_t)k * 324 + j], s);
    bbox[n * 324 + j] = s + bb[j];
  }
}

// ---------------------------------------------------------------------------
// 5) Row-wise softmax over 81 classes
// ---------------------------------------------------------------------------
__global__ void __launch_bounds__(128) softmax_kernel(const float* __restrict__ score,
                                                      float* __restrict__ prob) {
  __shared__ float red[128];
  const int n = blockIdx.x;
  const int t = threadIdx.x;
  const float v = (t < 81) ? score[n * 81 + t] : -1e30f;
  red[t] = v;
  __syncthreads();
  for (int s = 64; s > 0; s >>= 1) {
    if (t < s) red[t] = fmaxf(red[t], red[t + s]);
    __syncthreads();
  }
  const float m = red[0];
  __syncthreads();
  const float e = (t < 81) ? expf(v - m) : 0.0f;
  red[t] = e;
  __syncthreads();
  for (int s = 64; s > 0; s >>= 1) {
    if (t < s) red[t] += red[t + s];
    __syncthreads();
  }
  if (t < 81) prob[n * 81 + t] = e / red[0];
}

// ---------------------------------------------------------------------------
extern "C" void kernel_launch(void* const* d_in, const int* in_sizes, int n_in,
                              void* d_out, int out_size, void* d_ws, size_t ws_size,
                              hipStream_t stream) {
  const float* fmap  = (const float*)d_in[0];  // [1,64,64,1024]
  const float* boxes = (const float*)d_in[1];  // [256,4]
  const float* Wres  = (const float*)d_in[2];  // [3,3,1024,2048]
  const float* bres  = (const float*)d_in[3];  // [2048]
  const float* Wc    = (const float*)d_in[4];  // [2048,81]
  const float* bc    = (const float*)d_in[5];  // [81]
  const float* Wb    = (const float*)d_in[6];  // [2048,324]
  const float* bb    = (const float*)d_in[7];  // [324]

  float* out   = (float*)d_out;
  float* score = out;                 // [256,81]
  float* prob  = out + 256 * 81;      // [256,81]
  float* bbox  = out + 2 * 256 * 81;  // [256,324]

  char* ws = (char*)d_ws;
  unsigned short* Wt   = (unsigned short*)ws;                                   // 2048*9216 bf16
  unsigned short* roif = (unsigned short*)(ws + (size_t)2048 * 9216 * 2);       // 256*49*1024 bf16
  float*          fc   = (float*)(ws + (size_t)2048 * 9216 * 2
                                     + (size_t)256 * 49 * 1024 * 2);            // 256*2048 f32

  wconv_kernel<<<dim3(64, 32, 9), 256, 0, stream>>>(Wres, Wt);
  crop_pool_kernel<<<dim3(49, 256), 256, 0, stream>>>(fmap, boxes, roif);
  conv_gemm_kernel<<<dim3(16, 256), 256, 0, stream>>>(roif, Wt, bres, fc);
  fc_kernel<<<256, 512, 0, stream>>>(fc, Wc, bc, Wb, bb, score, bbox);
  softmax_kernel<<<256, 128, 0, stream>>>(score, prob);
}
